// VectorQuantizer_61297773248841
// MI455X (gfx1250) — matrix operand order repaired and checked by reading, not compiled, not run on hardware
//
#include <hip/hip_runtime.h>

typedef __attribute__((ext_vector_type(16))) _Float16 v16h;
typedef __attribute__((ext_vector_type(8)))  float    v8f;

#define N_CODES 1024
#define DIM     64
#define HW      4096            // 64*64 per batch image
#define BATCH   32
#define NPOS    (BATCH * HW)    // 131072 positions
#define TOTAL   (NPOS * DIM)    // 8388608 elements of z / z_q

union AFrag { uint4 u[2]; v16h h; };

// ---------------------------------------------------------------------------
// Prep: codebook fp32 -> fp16 (for WMMA), per-code -0.5*|e|^2, zero loss slot.
// ---------------------------------------------------------------------------
__global__ void vq_prep(const float* __restrict__ cb,
                        _Float16* __restrict__ cb16,
                        float* __restrict__ hns,
                        float* __restrict__ loss_slot) {
    int j = blockIdx.x * blockDim.x + threadIdx.x;
    if (j == 0) *loss_slot = 0.0f;
    if (j < N_CODES) {
        const float* row = cb + j * DIM;
        _Float16* orow = cb16 + j * DIM;
        float s = 0.0f;
#pragma unroll
        for (int c = 0; c < DIM; ++c) {
            float v = row[c];
            s += v * v;
            orow[c] = (_Float16)v;
        }
        hns[j] = -0.5f * s;   // folded into WMMA accumulator init
    }
}

// ---------------------------------------------------------------------------
// Main: each wave owns 32 positions (two 16-wide D tiles sharing every A
// fragment and hn init). Scores vs all 1024 codes via V_WMMA_F32_16X16X32_F16
// with per-slot running maxima (8 independent cmp/cndmask chains per tile).
// ---------------------------------------------------------------------------
__global__ void __launch_bounds__(256)
vq_main(const float* __restrict__ z,
        const float* __restrict__ cb,
        const _Float16* __restrict__ cb16,
        const float* __restrict__ hns,
        float* __restrict__ out,
        float* __restrict__ loss) {
    extern __shared__ __align__(16) char smem[];
    _Float16* sCB = (_Float16*)smem;                          // 1024*64 f16 = 128KB
    float*    sHN = (float*)(smem + N_CODES * DIM * 2);       // 1024 f32  =   4KB

    // Cooperative stage of codebook (f16) + half-norms into LDS.
    {
        const uint4* src = (const uint4*)cb16;
        uint4* dst = (uint4*)sCB;
        const int nvec = N_CODES * DIM * 2 / 16;              // 8192 x 16B
        for (int i = threadIdx.x; i < nvec; i += blockDim.x) dst[i] = src[i];
        for (int i = threadIdx.x; i < N_CODES; i += blockDim.x) sHN[i] = hns[i];
    }
    __syncthreads();

    const int wave = threadIdx.x >> 5;          // 8 waves / workgroup
    const int lane = threadIdx.x & 31;
    const int pos0 = (blockIdx.x * 8 + wave) * 32;   // 32 consecutive positions
    const int b    = pos0 >> 12;                // batch (32 | 4096: never crosses)
    const int hw0  = pos0 & 4095;

    const int col   = lane & 15;                // position within subtile
    const int ksel  = lane >> 4;                // lane-half selector
    const int row16 = lane & 15;                // code row within A tile

    // ---- B fragments for both subtiles (K=64 split 2x32), f32 -> f16 -------
    // 16-bit B 32x16 layout: dword j holds K=8h+2j,+1 (j<4), 16+8h+2(j-4),+1
    // (j>=4), h=lane/16. z is NCHW: channel stride HW, positions contiguous.
    v16h bf0a, bf1a, bf0b, bf1b;
    const float* zb = z + (size_t)b * DIM * HW + hw0 + col;
#pragma unroll
    for (int j = 0; j < 8; ++j) {
        int c0 = ksel * 8 + 2 * j + (j >= 4 ? 8 : 0);
        int c1 = c0 + 32;
        bf0a[2 * j]     = (_Float16)zb[(size_t)(c0)     * HW];
        bf0a[2 * j + 1] = (_Float16)zb[(size_t)(c0 + 1) * HW];
        bf1a[2 * j]     = (_Float16)zb[(size_t)(c1)     * HW];
        bf1a[2 * j + 1] = (_Float16)zb[(size_t)(c1 + 1) * HW];
        bf0b[2 * j]     = (_Float16)zb[(size_t)(c0)     * HW + 16];
        bf0b[2 * j + 1] = (_Float16)zb[(size_t)(c0 + 1) * HW + 16];
        bf1b[2 * j]     = (_Float16)zb[(size_t)(c1)     * HW + 16];
        bf1b[2 * j + 1] = (_Float16)zb[(size_t)(c1 + 1) * HW + 16];
    }

    // Per-slot running maxima: slot r <-> accumulator VGPR r (independent chains)
    float bsA[8], bsB[8];
    int   btA[8], btB[8];
#pragma unroll
    for (int r = 0; r < 8; ++r) {
        bsA[r] = -3.0e38f; bsB[r] = -3.0e38f; btA[r] = 0; btB[r] = 0;
    }

    const float* hbase = sHN + ksel * 8;
    for (int j0 = 0; j0 < N_CODES; j0 += 16) {
        // C init: -0.5|e|^2 for codes j0 + ksel*8 + 0..7 (shared by both tiles)
        const float4* hp = (const float4*)(hbase + j0);
        float4 h0 = hp[0], h1 = hp[1];
        v8f acc, acc2;
        acc[0] = h0.x; acc[1] = h0.y; acc[2] = h0.z; acc[3] = h0.w;
        acc[4] = h1.x; acc[5] = h1.y; acc[6] = h1.z; acc[7] = h1.w;
        acc2 = acc;

        // A fragments from LDS (16-bit A 16x32 layout), shared by both tiles.
        const _Float16* arow = sCB + (size_t)(j0 + row16) * DIM;
        AFrag a0, a1;
        a0.u[0] = *(const uint4*)(arow +      ksel * 8);
        a0.u[1] = *(const uint4*)(arow + 16 + ksel * 8);
        a1.u[0] = *(const uint4*)(arow + 32 + ksel * 8);
        a1.u[1] = *(const uint4*)(arow + 48 + ksel * 8);

        acc  = __builtin_amdgcn_wmma_f32_16x16x32_f16(
                   false, a0.h, false, bf0a, (short)0, acc,  false, false);
        acc2 = __builtin_amdgcn_wmma_f32_16x16x32_f16(
                   false, a0.h, false, bf0b, (short)0, acc2, false, false);
        acc  = __builtin_amdgcn_wmma_f32_16x16x32_f16(
                   false, a1.h, false, bf1a, (short)0, acc,  false, false);
        acc2 = __builtin_amdgcn_wmma_f32_16x16x32_f16(
                   false, a1.h, false, bf1b, (short)0, acc2, false, false);

#pragma unroll
        for (int r = 0; r < 8; ++r) {
            float sa = acc[r];
            if (sa > bsA[r]) { bsA[r] = sa; btA[r] = j0; }
            float sb = acc2[r];
            if (sb > bsB[r]) { bsB[r] = sb; btB[r] = j0; }
        }
    }

    // ---- Resolve per-slot maxima to (score, code), then cross-half combine --
    float bestA = -3.0e38f, bestB = -3.0e38f;
    int   codeA = 0x7fffffff, codeB = 0x7fffffff;
#pragma unroll
    for (int r = 0; r < 8; ++r) {
        int ca = btA[r] + ksel * 8 + r;
        if (bsA[r] > bestA || (bsA[r] == bestA && ca < codeA)) { bestA = bsA[r]; codeA = ca; }
        int cbx = btB[r] + ksel * 8 + r;
        if (bsB[r] > bestB || (bsB[r] == bestB && cbx < codeB)) { bestB = bsB[r]; codeB = cbx; }
    }
    {
        float os = __shfl_xor(bestA, 16, 32);
        int   oc = __shfl_xor(codeA, 16, 32);
        if (os > bestA || (os == bestA && oc < codeA)) { bestA = os; codeA = oc; }
        os = __shfl_xor(bestB, 16, 32);
        oc = __shfl_xor(codeB, 16, 32);
        if (os > bestB || (os == bestB && oc < codeB)) { bestB = os; codeB = oc; }
    }

    // ---- Output: z_q (NCHW, coalesced) + loss partial -----------------------
    const int p      = lane & 15;
    const int chbase = (lane >> 4) * 32;        // lanes<16: c 0..31, else 32..63
    const float* crowA = cb + (size_t)codeA * DIM;   // fp32 codebook rows
    const float* crowB = cb + (size_t)codeB * DIM;
    const float* zcol  = z   + (size_t)b * DIM * HW + hw0 + p;
    float*       ocol  = out + (size_t)b * DIM * HW + hw0 + p;
    float lsum = 0.0f;
#pragma unroll
    for (int c = 0; c < 32; ++c) {
        int ch = chbase + c;
        size_t off = (size_t)ch * HW;
        float ea = crowA[ch];
        float da = ea - zcol[off];
        lsum += da * da;
        ocol[off] = ea;                          // subtile a: position hw0+p
        float eb = crowB[ch];
        float db = eb - zcol[off + 16];
        lsum += db * db;
        ocol[off + 16] = eb;                     // subtile b: position hw0+16+p
    }
#pragma unroll
    for (int off = 16; off > 0; off >>= 1) lsum += __shfl_down(lsum, off, 32);
    if (lane == 0) atomicAdd(loss, lsum * (1.25f / (float)TOTAL));
}

// ---------------------------------------------------------------------------
extern "C" void kernel_launch(void* const* d_in, const int* in_sizes, int n_in,
                              void* d_out, int out_size, void* d_ws, size_t ws_size,
                              hipStream_t stream) {
    (void)in_sizes; (void)n_in; (void)out_size; (void)ws_size;
    const float* z  = (const float*)d_in[0];
    const float* cb = (const float*)d_in[1];
    float* out = (float*)d_out;                 // [TOTAL] z_q_st, then [1] vq_loss
    float* loss = out + TOTAL;

    _Float16* cb16 = (_Float16*)d_ws;                                 // 128KB
    float*    hns  = (float*)((char*)d_ws + N_CODES * DIM * sizeof(_Float16));

    vq_prep<<<4, 256, 0, stream>>>(cb, cb16, hns, loss);

    const size_t lds = (size_t)N_CODES * DIM * 2 + N_CODES * 4;       // 132KB
    vq_main<<<NPOS / 256, 256, lds, stream>>>(z, cb, cb16, hns, out, loss);
}
